// VxNet_34479997452376
// MI455X (gfx1250) — matrix operand order repaired and checked by reading, not compiled, not run on hardware
//
#include <hip/hip_runtime.h>
#include <hip/hip_bf16.h>

typedef __attribute__((ext_vector_type(16))) _Float16 v16h;
typedef __attribute__((ext_vector_type(8)))  _Float16 v8h;
typedef __attribute__((ext_vector_type(8)))  float    v8f;
typedef __attribute__((ext_vector_type(4)))  unsigned int uint4v;

// ---------------------------------------------------------------- utilities

__global__ void zero_u4_kernel(uint4v* __restrict__ p, long n) {
  long i = (long)blockIdx.x * blockDim.x + threadIdx.x;
  if (i < n) { uint4v z = {0u, 0u, 0u, 0u}; p[i] = z; }
}

__global__ void scatter_kernel(const float* __restrict__ vf, const int* __restrict__ coors,
                               _Float16* __restrict__ act, float* __restrict__ mask,
                               int N, int H, int W) {
  int i = blockIdx.x * blockDim.x + threadIdx.x;
  if (i >= N) return;
  int z = coors[i*4 + 1], y = coors[i*4 + 2], x = coors[i*4 + 3];
  size_t vox = ((size_t)z * H + y) * W + x;
  mask[vox] = 1.0f;
#pragma unroll
  for (int c = 0; c < 4; ++c) act[vox*16 + c] = (_Float16)vf[i*4 + c];
}

__global__ void bn_prep_kernel(const float* __restrict__ g, const float* __restrict__ b,
                               const float* __restrict__ m, const float* __restrict__ v,
                               float* __restrict__ scale, float* __restrict__ bias, int C) {
  int i = blockIdx.x * blockDim.x + threadIdx.x;
  if (i >= C) return;
  float s = g[i] * rsqrtf(v[i] + 1e-3f);
  scale[i] = s;
  bias[i]  = b[i] - m[i] * s;
}

// Pack f32 weights (kd,kh,kw,cin,cout) into the exact WMMA-B f16 lane layout:
// element (ntile, kc, lane, e):  N = ntile*16 + lane%16,  K = kc*32 + (e%8) + 8*(lane/16) + 16*(e/8)
__global__ void pack_b_kernel(const float* __restrict__ W, _Float16* __restrict__ Bp,
                              int CIN, int CPAD, int COUT, int NTAPS, int KC) {
  long idx = (long)blockIdx.x * blockDim.x + threadIdx.x;
  long total = (long)(COUT >> 4) * KC * 512;
  if (idx >= total) return;
  int e    = (int)(idx & 15);
  int lane = (int)((idx >> 4) & 31);
  long q   = idx >> 9;                 // ntile*KC + kc
  int kc    = (int)(q % KC);
  int ntile = (int)(q / KC);
  int n = ntile*16 + (lane & 15);
  int k = kc*32 + (e & 7) + ((lane >> 4) << 3) + ((e >> 3) << 4);
  int tap = k / CPAD, c = k % CPAD;
  float val = 0.0f;
  if (tap < NTAPS && c < CIN) val = W[((size_t)tap * CIN + c) * COUT + n];
  Bp[idx] = (_Float16)val;
}

__global__ void pool_mask_kernel(const float* __restrict__ mi, float* __restrict__ mo,
                                 int Din, int Hin, int Win, int Dout, int Hout, int Wout) {
  long i = (long)blockIdx.x * blockDim.x + threadIdx.x;
  long nv = (long)Dout * Hout * Wout;
  if (i >= nv) return;
  int x = (int)(i % Wout); long t = i / Wout;
  int y = (int)(t % Hout); int z = (int)(t / Hout);
  float m = 0.0f;
  for (int dz = 0; dz < 3; ++dz) { int iz = 2*z + dz - 1; if ((unsigned)iz >= (unsigned)Din) continue;
    for (int dy = 0; dy < 3; ++dy) { int iy = 2*y + dy - 1; if ((unsigned)iy >= (unsigned)Hin) continue;
      for (int dx = 0; dx < 3; ++dx) { int ix = 2*x + dx - 1; if ((unsigned)ix >= (unsigned)Win) continue;
        m = fmaxf(m, mi[((size_t)iz * Hin + iy) * Win + ix]);
      } } }
  mo[i] = m;
}

// ---------------------------------------------------------------- implicit-GEMM conv
//
// D = A(16 voxels x K) * B(K x 16*NT couts), K = NTAPS*CPAD padded to 32.
// One wave per voxel-tile; each wave carries NT cout-tile accumulators so one
// guarded A gather feeds NT independent WMMAs. 4 waves/block share the B tiles,
// staged through LDS in segments of SEG chunks (NT*SEG KB <= 28 KB).
// All voxel counts are multiples of 64 -> wave-uniform control flow, EXEC all-1s
// at every WMMA, and barriers are safe.
constexpr int pick_seg(int KC, int NT) {
  int best = 1;
  for (int s = 1; s <= KC; ++s)
    if (KC % s == 0 && NT * s * 1024 <= 28672) best = s;
  return best;
}

template<int CPAD, int COUT, int NT, int NTAPS, int STRIDE>
__global__ __launch_bounds__(128) void conv_bnsilu(
    const _Float16* __restrict__ in, const _Float16* __restrict__ Bpack,
    const float* __restrict__ bnscale, const float* __restrict__ bnbias,
    const float* __restrict__ mask_out,
    _Float16* __restrict__ out, float* __restrict__ m_out,
    int Din, int Hin, int Win, int Dout, int Hout, int Wout)
{
  constexpr int KC   = (NTAPS * CPAD + 31) / 32;   // K chunks of 32
  constexpr int SEG  = pick_seg(KC, NT);           // chunks staged per segment
  constexpr int NSEG = KC / SEG;
  static_assert(SEG * NSEG == KC, "segments must tile KC exactly");

  __shared__ __align__(64) _Float16 Bs[NT * SEG * 512];

  const int lane  = threadIdx.x & 31;
  const int half  = lane >> 4;
  const int wid   = threadIdx.x >> 5;
  const long tile = (long)blockIdx.x * 4 + wid;
  const int ntb   = blockIdx.y * NT;               // first cout-tile of this wave

  // ---- per-lane A-row geometry (ISA A layout: M = lane%16, half selects K+8) ----
  const long vox = tile * 16 + (lane & 15);
  int xo = (int)(vox % Wout); long t0 = vox / Wout;
  int yo = (int)(t0 % Hout);  int zo = (int)(t0 / Hout);

  const char* inB  = (const char*)in;
  const int  cbyte = 16 * half;                    // c0 = 8*half, in bytes
  bool vz[3] = {false,false,false}, vy[3] = {false,false,false}, vx[3] = {false,false,false};
  int  bzB[3] = {0,0,0}, byB[3] = {0,0,0}, bxB[3] = {0,0,0};  // byte offsets (i32)
  if (NTAPS == 27) {
#pragma unroll
    for (int d = 0; d < 3; ++d) {
      int iz = zo * STRIDE + d - 1; vz[d] = (unsigned)iz < (unsigned)Din; bzB[d] = iz * (Hin * Win * CPAD * 2);
      int iy = yo * STRIDE + d - 1; vy[d] = (unsigned)iy < (unsigned)Hin; byB[d] = iy * (Win * CPAD * 2);
      int ix = xo * STRIDE + d - 1; vx[d] = (unsigned)ix < (unsigned)Win; bxB[d] = ix * (CPAD * 2) + cbyte;
    }
  }
  const int off11 = (int)(((((size_t)zo * Hin + yo) * Win + xo)) * (CPAD * 2)) + cbyte;

  v8f acc[NT];
#pragma unroll
  for (int nt = 0; nt < NT; ++nt) acc[nt] = (v8f){0.f,0.f,0.f,0.f,0.f,0.f,0.f,0.f};

#pragma unroll
  for (int seg = 0; seg < NSEG; ++seg) {
    // cooperative stage of this segment's NT B tiles into LDS
    for (int i = threadIdx.x; i < NT * SEG * 64; i += 128) {
      int nt = i / (SEG * 64);
      int j  = i - nt * (SEG * 64);
      *(v8h*)(Bs + (size_t)i * 8) =
          *(const v8h*)(Bpack + ((size_t)(ntb + nt) * KC + seg * SEG) * 512 + (size_t)j * 8);
    }
    __syncthreads();
    if (seg + 1 < NSEG)
      __builtin_prefetch(Bpack + ((size_t)ntb * KC + (seg + 1) * (size_t)SEG) * 512 +
                         (size_t)threadIdx.x * 8, 0, 0);

#pragma unroll
    for (int kcl = 0; kcl < SEG; ++kcl) {
      const int kc = seg * SEG + kcl;              // compile-time after unroll
      v8h g0 = {}, g1 = {};
      if (NTAPS == 1) {
        g0 = *(const v8h*)(inB + off11 + kc * 64);
        g1 = *(const v8h*)(inB + off11 + kc * 64 + 32);
      } else if (CPAD == 16) {
        // one chunk covers two taps; c0 = 8*half folded into bxB
        const int tap0 = 2 * kc, tap1 = 2 * kc + 1;
        {
          const int dz = tap0 / 9, r = tap0 % 9, dy = r / 3, dx = r % 3;
          if (vz[dz] && vy[dy] && vx[dx])
            g0 = *(const v8h*)(inB + (bzB[dz] + byB[dy] + bxB[dx]));
        }
        if (tap1 < NTAPS) {
          const int dz = tap1 / 9, r = tap1 % 9, dy = r / 3, dx = r % 3;
          if (vz[dz] && vy[dy] && vx[dx])
            g1 = *(const v8h*)(inB + (bzB[dz] + byB[dy] + bxB[dx]));
        }
      } else {
        // CPAD >= 32: whole chunk inside one tap; groups at +0 / +32 bytes
        const int tap = (kc * 32) / CPAD;
        const int cb  = ((kc * 32) % CPAD) * 2;    // literal instruction offset
        const int dz = tap / 9, r = tap % 9, dy = r / 3, dx = r % 3;
        if (vz[dz] && vy[dy] && vx[dx]) {
          const char* p = inB + (bzB[dz] + byB[dy] + bxB[dx]);
          g0 = *(const v8h*)(p + cb);
          g1 = *(const v8h*)(p + cb + 32);
        }
      }
      v16h a = __builtin_shufflevector(g0, g1, 0,1,2,3,4,5,6,7,8,9,10,11,12,13,14,15);
#pragma unroll
      for (int nt = 0; nt < NT; ++nt) {            // NT independent WMMAs per A gather
        v16h b = *(const v16h*)(Bs + ((size_t)(nt * SEG + kcl) * 512) + (size_t)lane * 16);
        acc[nt] = __builtin_amdgcn_wmma_f32_16x16x32_f16(false, a, false, b, (short)0,
                                                         acc[nt], false, false);
      }
    }
    if (seg + 1 < NSEG) __syncthreads();
  }

  // ---- fused epilogue: BN + SiLU + mask ----
  // D layout: lane -> N = lane%16 ; vgpr v -> M = v + 8*half.
  // The 8 voxels vob..vob+7 stay in one W-row (vob and W are multiples of 8).
  const long vob = tile * 16 + 8 * half;
  int xb = (int)(vob % Wout); long tb = vob / Wout;
  int yb = (int)(tb % Hout);  int zb = (int)(tb / Hout);
  float mk[8];
#pragma unroll
  for (int v = 0; v < 8; ++v) mk[v] = mask_out[vob + v];
#pragma unroll
  for (int nt = 0; nt < NT; ++nt) {
    const int co = (ntb + nt) * 16 + (lane & 15);
    const float sc = bnscale[co];
    const float bi = bnbias[co];
    _Float16* orow = out + (size_t)vob * COUT + co;
    float* mrow = m_out ? m_out + (((size_t)co * Dout + zb) * Hout + yb) * Wout + xb : nullptr;
#pragma unroll
    for (int v = 0; v < 8; ++v) {
      float y = acc[nt][v] * sc + bi;
      float s = y / (1.0f + __expf(-y));          // SiLU
      float r = s * mk[v];
      orow[(size_t)v * COUT] = (_Float16)r;
      if (m_out) mrow[v] = r;
    }
  }
}

template<int CPAD, int COUT, int NT, int NTAPS, int STRIDE>
static void run_conv(const void* in, const void* bp, const float* sc, const float* bi,
                     const float* mask, void* out, float* mout,
                     int Din, int Hin, int Win, int Dout, int Hout, int Wout,
                     hipStream_t s) {
  long nv = (long)Dout * Hout * Wout;
  long tiles = (nv + 15) >> 4;                    // all exact multiples of 4 tiles/block
  dim3 grid((unsigned)((tiles + 3) >> 2), COUT / (16 * NT));
  conv_bnsilu<CPAD, COUT, NT, NTAPS, STRIDE><<<grid, dim3(128), 0, s>>>(
      (const _Float16*)in, (const _Float16*)bp, sc, bi, mask,
      (_Float16*)out, mout, Din, Hin, Win, Dout, Hout, Wout);
}

// ---------------------------------------------------------------- host driver

extern "C" void kernel_launch(void* const* d_in, const int* in_sizes, int n_in,
                              void* d_out, int out_size, void* d_ws, size_t ws_size,
                              hipStream_t stream) {
  (void)n_in; (void)out_size; (void)ws_size;
  const float* vf    = (const float*)d_in[0];
  const int*   coors = (const int*)d_in[1];
  const int    NVOX  = in_sizes[0] / 4;
  auto P = [&](int layer, int j) { return (const float*)d_in[3 + layer*5 + j]; };

  // resolutions
  const int D0=41, H0=320, W0=320;
  const int D1=21, H1=160, W1=160;
  const int D2=11, H2=80,  W2=80;
  const int D3=6,  H3=40,  W3=40;
  const size_t NV0 = (size_t)D0*H0*W0, NV1 = (size_t)D1*H1*W1;
  const size_t NV2 = (size_t)D2*H2*W2, NV3 = (size_t)D3*H3*W3;

  // layer table: cin, cpad_in, cout, ntaps
  struct LC { int cin, cpad, cout, ntaps; };
  static const LC L[14] = {
    {4,16,16,27},  {16,16,16,27}, {16,16,16,27}, {16,16,16,27}, {16,16,16,27},
    {16,16,32,27}, {32,32,64,27}, {64,64,64,27}, {64,64,64,27}, {64,64,128,27},
    {128,128,128,27}, {128,128,128,27}, {128,128,128,27}, {128,128,256,1} };

  // workspace layout
  char* ws = (char*)d_ws;
  size_t ofs = 0;
  auto alloc = [&](size_t b) { size_t o = ofs; ofs += (b + 255) & ~(size_t)255; return o; };
  const size_t actBytes = NV0 * 16 * sizeof(_Float16);        // 134.3 MB each
  size_t oA0 = alloc(actBytes), oA1 = alloc(actBytes);
  size_t oM0 = alloc(NV0*4), oM1 = alloc(NV1*4), oM2 = alloc(NV2*4), oM3 = alloc(NV3*4);
  size_t oBN = alloc(14 * 512 * sizeof(float));
  size_t oBP[14];
  int    KCv[14];
  for (int i = 0; i < 14; ++i) {
    KCv[i]  = (L[i].ntaps * L[i].cpad + 31) / 32;
    oBP[i]  = alloc((size_t)L[i].cout * KCv[i] * 64);          // (COUT/16)*KC*512 halfs
  }

  _Float16* A0 = (_Float16*)(ws + oA0);
  _Float16* A1 = (_Float16*)(ws + oA1);
  float* M0 = (float*)(ws + oM0); float* M1 = (float*)(ws + oM1);
  float* M2 = (float*)(ws + oM2); float* M3 = (float*)(ws + oM3);
  auto SC = [&](int i) { return (float*)(ws + oBN) + i*512; };
  auto BI = [&](int i) { return (float*)(ws + oBN) + i*512 + 256; };
  auto BP = [&](int i) { return (void*)(ws + oBP[i]); };

  // zero dense input + mask (channel padding stays zero)
  { long n = (long)(actBytes / 16);
    zero_u4_kernel<<<dim3((unsigned)((n + 255)/256)), dim3(256), 0, stream>>>((uint4v*)A0, n); }
  { long n = (long)(NV0 * 4 / 16);
    zero_u4_kernel<<<dim3((unsigned)((n + 255)/256)), dim3(256), 0, stream>>>((uint4v*)M0, n); }

  // per-layer BN fold + weight packing into WMMA-B layout
  for (int i = 0; i < 14; ++i) {
    bn_prep_kernel<<<dim3((L[i].cout + 63)/64), dim3(64), 0, stream>>>(
        P(i,1), P(i,2), P(i,3), P(i,4), SC(i), BI(i), L[i].cout);
    long total = (long)(L[i].cout >> 4) * KCv[i] * 512;
    pack_b_kernel<<<dim3((unsigned)((total + 255)/256)), dim3(256), 0, stream>>>(
        P(i,0), (_Float16*)BP(i), L[i].cin, L[i].cpad, L[i].cout, L[i].ntaps, KCv[i]);
  }

  // scatter sparse voxels into dense f16 grid + mask
  scatter_kernel<<<dim3((NVOX + 255)/256), dim3(256), 0, stream>>>(
      vf, coors, A0, M0, NVOX, H0, W0);

  float* out_f = (float*)d_out;
  const size_t M1OFF = 0;
  const size_t M2OFF = (size_t)16 * NV1;            // 8,601,600
  const size_t M3OFF = M2OFF + (size_t)64 * NV2;    // 13,107,200

  // ---- full resolution ----
  run_conv<16,16,1,27,1>(A0, BP(0),  SC(0),  BI(0),  M0, A1, nullptr, D0,H0,W0, D0,H0,W0, stream);
  run_conv<16,16,1,27,1>(A1, BP(1),  SC(1),  BI(1),  M0, A0, nullptr, D0,H0,W0, D0,H0,W0, stream);
  // ---- down 1 ----
  pool_mask_kernel<<<dim3((unsigned)((NV1 + 255)/256)), dim3(256), 0, stream>>>(
      M0, M1, D0,H0,W0, D1,H1,W1);
  run_conv<16,16,1,27,2>(A0, BP(2),  SC(2),  BI(2),  M1, A1, nullptr, D0,H0,W0, D1,H1,W1, stream);
  run_conv<16,16,1,27,1>(A1, BP(3),  SC(3),  BI(3),  M1, A0, nullptr, D1,H1,W1, D1,H1,W1, stream);
  run_conv<16,16,1,27,1>(A0, BP(4),  SC(4),  BI(4),  M1, A1, out_f + M1OFF,
                         D1,H1,W1, D1,H1,W1, stream);                       // m1
  // ---- down 2 ----
  pool_mask_kernel<<<dim3((unsigned)((NV2 + 255)/256)), dim3(256), 0, stream>>>(
      M1, M2, D1,H1,W1, D2,H2,W2);
  run_conv<16,32,2,27,2>(A1, BP(5),  SC(5),  BI(5),  M2, A0, nullptr, D1,H1,W1, D2,H2,W2, stream);
  run_conv<32,64,2,27,1>(A0, BP(6),  SC(6),  BI(6),  M2, A1, nullptr, D2,H2,W2, D2,H2,W2, stream);
  run_conv<64,64,2,27,1>(A1, BP(7),  SC(7),  BI(7),  M2, A0, nullptr, D2,H2,W2, D2,H2,W2, stream);
  run_conv<64,64,2,27,1>(A0, BP(8),  SC(8),  BI(8),  M2, A1, out_f + M2OFF,
                         D2,H2,W2, D2,H2,W2, stream);                       // m2
  // ---- down 3 ----
  pool_mask_kernel<<<dim3((unsigned)((NV3 + 255)/256)), dim3(256), 0, stream>>>(
      M2, M3, D2,H2,W2, D3,H3,W3);
  run_conv<64,128,4,27,2>(A1, BP(9),  SC(9),  BI(9),  M3, A0, nullptr, D2,H2,W2, D3,H3,W3, stream);
  run_conv<128,128,4,27,1>(A0, BP(10), SC(10), BI(10), M3, A1, nullptr, D3,H3,W3, D3,H3,W3, stream);
  run_conv<128,128,4,27,1>(A1, BP(11), SC(11), BI(11), M3, A0, nullptr, D3,H3,W3, D3,H3,W3, stream);
  run_conv<128,128,4,27,1>(A0, BP(12), SC(12), BI(12), M3, A1, nullptr, D3,H3,W3, D3,H3,W3, stream);
  // ---- 1x1x1 head ----
  run_conv<128,256,4,1,1>(A1, BP(13), SC(13), BI(13), M3, A0, out_f + M3OFF,
                          D3,H3,W3, D3,H3,W3, stream);                      // m3
}
